// HeteroNet_41102837023556
// MI455X (gfx1250) — compile-verified, matrix operand order checked by reading, not run on hardware
//
#include <hip/hip_runtime.h>

// ---------- types ----------
typedef __attribute__((ext_vector_type(16))) __bf16 v16bf;
typedef __attribute__((ext_vector_type(8)))  __bf16 bf16x8;
typedef __attribute__((ext_vector_type(8)))  float  v8f;
typedef __attribute__((ext_vector_type(4)))  float  f32x4;

union V16U { v16bf v; bf16x8 h[2]; };

#define NEG_SLOPE 0.01f
#define FEAT 512

// ---------- elementwise: leaky-relu f32 -> bf16 (8 elems / thread) ----------
__global__ void k_lrelu_bf16(const float* __restrict__ x, __bf16* __restrict__ y, long n8) {
    long i = (long)blockIdx.x * blockDim.x + threadIdx.x;
    if (i >= n8) return;
    const f32x4* p = (const f32x4*)x + i * 2;
    f32x4 a = p[0], b = p[1];
    bf16x8 o;
#pragma unroll
    for (int j = 0; j < 4; ++j) {
        float va = a[j]; va = va > 0.f ? va : va * NEG_SLOPE; o[j]     = (__bf16)va;
        float vb = b[j]; vb = vb > 0.f ? vb : vb * NEG_SLOPE; o[4 + j] = (__bf16)vb;
    }
    *(bf16x8*)(y + i * 8) = o;
}

// ---------- weight convert: W[fin,fout] f32 -> WT[fout,fin] bf16 ----------
__global__ void k_wt_bf16(const float* __restrict__ W, __bf16* __restrict__ WT,
                          int fin, int fout) {
    long idx = (long)blockIdx.x * blockDim.x + threadIdx.x;
    if (idx >= (long)fin * fout) return;
    int o = (int)(idx / fin);
    int i = (int)(idx % fin);
    WT[idx] = (__bf16)W[(long)i * fout + o];
}

// ---------- zero f32 buffer (4 elems / thread) ----------
__global__ void k_zero_f32(float* __restrict__ p, long n4) {
    long i = (long)blockIdx.x * blockDim.x + threadIdx.x;
    if (i >= n4) return;
    f32x4 z = {};
    *((f32x4*)p + i) = z;
}

// ---------- edge degree count ----------
__global__ void k_count(const int* __restrict__ dst, float* __restrict__ cnt, int E) {
    int e = blockIdx.x * blockDim.x + threadIdx.x;
    if (e >= E) return;
    __hip_atomic_fetch_add(cnt + dst[e], 1.0f, __ATOMIC_RELAXED, __HIP_MEMORY_SCOPE_AGENT);
}

// ---------- scatter-add of gathered bf16 rows into f32 accumulator ----------
// one thread handles 8 contiguous features of one edge
__global__ void k_scatter(const __bf16* __restrict__ feat, const int* __restrict__ ei,
                          float* __restrict__ agg, int E) {
    long idx = (long)blockIdx.x * blockDim.x + threadIdx.x;
    if (idx >= (long)E * (FEAT / 8)) return;
    int e = (int)(idx >> 6);
    int c = (int)(idx & 63) * 8;
    int s = ei[e];
    int d = ei[E + e];
    bf16x8 v = *(const bf16x8*)(feat + (long)s * FEAT + c);
    float* q = agg + (long)d * FEAT + c;
#pragma unroll
    for (int j = 0; j < 8; ++j)
        __hip_atomic_fetch_add(q + j, (float)v[j], __ATOMIC_RELAXED, __HIP_MEMORY_SCOPE_AGENT);
}

// ---------- normalize by degree, f32 -> bf16 ----------
__global__ void k_norm_bf16(const float* __restrict__ agg, const float* __restrict__ cnt,
                            __bf16* __restrict__ out, long rows) {
    long idx = (long)blockIdx.x * blockDim.x + threadIdx.x;
    if (idx >= rows * (FEAT / 8)) return;
    long row = idx >> 6;
    int  c   = (int)(idx & 63) * 8;
    float inv = 1.0f / fmaxf(cnt[row], 1.0f);
    const f32x4* p = (const f32x4*)(agg + row * FEAT + c);
    f32x4 a = p[0], b = p[1];
    bf16x8 o;
#pragma unroll
    for (int j = 0; j < 4; ++j) {
        o[j]     = (__bf16)(a[j] * inv);
        o[4 + j] = (__bf16)(b[j] * inv);
    }
    *(bf16x8*)(out + row * FEAT + c) = o;
}

// ---------- WMMA bf16 GEMM: C[M,512] = A[M,K] @ BT[512,K]^T + bias ----------
// block tile 128x128, K-step 32; 8 waves: wave = (wn<<2)|wm, wm -> 32 rows, wn -> 64 cols
// MODE 0: f32 out, 1: bf16 out, 2: leaky-relu + bf16 out
#define TM 128
#define TN 128
#define TK 32
#define LDSTR 40   // bf16 elems per LDS row (32 + 8 pad -> 80B, conflict-free frag reads)

template <int MODE>
__global__ __launch_bounds__(256)
void k_gemm_wmma(const __bf16* __restrict__ A, const __bf16* __restrict__ BT,
                 const float* __restrict__ bias, void* __restrict__ Cout,
                 int M, int K, int ldc, int colOff) {
    __shared__ __bf16 As[TM * LDSTR];
    __shared__ __bf16 Bs[TN * LDSTR];

    const int t     = threadIdx.x;
    const int lane  = t & 31;
    const int wave  = t >> 5;
    const int wm    = wave & 3;
    const int wn    = wave >> 2;
    const int mBase = blockIdx.x * TM;
    const int nBase = blockIdx.y * TN;
    const int lhalf = lane >> 4;   // K-half select per ISA 16-bit A/B layout
    const int l15   = lane & 15;

    v8f zeroV = {};
    v8f acc[2][4];
#pragma unroll
    for (int mf = 0; mf < 2; ++mf)
#pragma unroll
        for (int nf = 0; nf < 4; ++nf) acc[mf][nf] = zeroV;

    const int kIters = K / TK;
    for (int kt = 0; kt < kIters; ++kt) {
        const int k0 = kt * TK;
        // cooperative tile load: 128 rows x 32 bf16, 16B chunks, 2 per thread each for A and B
#pragma unroll
        for (int i = 0; i < 2; ++i) {
            int idx = t + i * 256;
            int row = idx >> 2;
            int ch  = (idx & 3) * 8;
            long gr = (long)(mBase + row);
            bf16x8 va = {};
            if (gr < M) {
                const __bf16* ap = A + gr * K + k0 + ch;
                va = *(const bf16x8*)ap;
                __builtin_prefetch(ap + TK, 0, 1);       // global_prefetch_b8 next K tile
            }
            *(bf16x8*)(As + row * LDSTR + ch) = va;
            const __bf16* bp = BT + (long)(nBase + row) * K + k0 + ch;
            *(bf16x8*)(Bs + row * LDSTR + ch) = *(const bf16x8*)bp;
            __builtin_prefetch(bp + TK, 0, 1);
        }
        __syncthreads();

        // fragment loads: lanes 0-15 take K[c..c+7]/K[c+16..c+23] with c=0, lanes 16-31 c=8
        V16U afrag[2], bfrag[4];
#pragma unroll
        for (int mf = 0; mf < 2; ++mf) {
            const __bf16* p = As + (wm * 32 + mf * 16 + l15) * LDSTR + lhalf * 8;
            afrag[mf].h[0] = *(const bf16x8*)p;
            afrag[mf].h[1] = *(const bf16x8*)(p + 16);
        }
#pragma unroll
        for (int nf = 0; nf < 4; ++nf) {
            const __bf16* p = Bs + (wn * 64 + nf * 16 + l15) * LDSTR + lhalf * 8;
            bfrag[nf].h[0] = *(const bf16x8*)p;
            bfrag[nf].h[1] = *(const bf16x8*)(p + 16);
        }
#pragma unroll
        for (int mf = 0; mf < 2; ++mf)
#pragma unroll
            for (int nf = 0; nf < 4; ++nf)
                acc[mf][nf] = __builtin_amdgcn_wmma_f32_16x16x32_bf16(
                    false, afrag[mf].v, false, bfrag[nf].v,
                    (short)0, acc[mf][nf], false, false);
        __syncthreads();
    }

    // epilogue: C/D layout -> lanes 0-15: M=r, lanes 16-31: M=r+8; N = lane&15
#pragma unroll
    for (int nf = 0; nf < 4; ++nf) {
        const int col = nBase + wn * 64 + nf * 16 + l15;
        const float b = bias[col];
#pragma unroll
        for (int mf = 0; mf < 2; ++mf) {
            const int rbase = mBase + wm * 32 + mf * 16 + lhalf * 8;
#pragma unroll
            for (int r = 0; r < 8; ++r) {
                const int row = rbase + r;
                if (row < M) {
                    float v = acc[mf][nf][r] + b;
                    if (MODE == 2) v = v > 0.f ? v : v * NEG_SLOPE;
                    long o = (long)row * ldc + colOff + col;
                    if (MODE == 0) ((float*)Cout)[o]   = v;
                    else           ((__bf16*)Cout)[o]  = (__bf16)v;
                }
            }
        }
    }
}

// ---------- host orchestration ----------
static inline int cdiv(long a, int b) { return (int)((a + b - 1) / b); }

extern "C" void kernel_launch(void* const* d_in, const int* in_sizes, int n_in,
                              void* d_out, int out_size, void* d_ws, size_t ws_size,
                              hipStream_t stream) {
    (void)n_in; (void)out_size; (void)ws_size;
    const int N0 = in_sizes[0] / FEAT;
    const int N1 = in_sizes[1] / FEAT;
    const int E  = in_sizes[2] / 2;
    const int NMAX = N0 > N1 ? N0 : N1;

    const float* x0  = (const float*)d_in[0];
    const float* x1  = (const float*)d_in[1];
    const int* ei01  = (const int*)d_in[2];
    const int* ei10  = (const int*)d_in[3];

    // workspace carve-up
    char* ws = (char*)d_ws;
    size_t off = 0;
    auto carve = [&](size_t bytes) -> char* {
        char* p = ws + off;
        off += (bytes + 255) & ~(size_t)255;
        return p;
    };
    const size_t WBLK = 1048576;                          // bf16 elems per sage (Wn+Ws+Wu)
    __bf16* WT    = (__bf16*)carve(4 * WBLK * sizeof(__bf16));
    __bf16* hb0   = (__bf16*)carve((size_t)N0 * FEAT * 2);
    __bf16* hb1   = (__bf16*)carve((size_t)N1 * FEAT * 2);
    __bf16* hb0n  = (__bf16*)carve((size_t)N0 * FEAT * 2);
    __bf16* hb1n  = (__bf16*)carve((size_t)N1 * FEAT * 2);
    __bf16* aggbf = (__bf16*)carve((size_t)NMAX * FEAT * 2);
    float*  cnt   = (float*)carve((size_t)NMAX * 4);
    char*   big   = carve((size_t)NMAX * FEAT * 4);        // aggf32 / hcat alias (disjoint lifetimes)
    float*  aggf  = (float*)big;
    __bf16* hcat  = (__bf16*)big;

    const int TB = 256;

    // leaky-relu inputs -> bf16
    {
        long n8 = (long)N0 * FEAT / 8;
        k_lrelu_bf16<<<cdiv(n8, TB), TB, 0, stream>>>(x0, hb0, n8);
        n8 = (long)N1 * FEAT / 8;
        k_lrelu_bf16<<<cdiv(n8, TB), TB, 0, stream>>>(x1, hb1, n8);
    }
    // weights -> transposed bf16 (sage s: 0=l1_01, 1=l1_10, 2=l2_01, 3=l2_10)
    for (int s = 0; s < 4; ++s) {
        const int base = 4 + s * 6;
        __bf16* wb = WT + (size_t)s * WBLK;
        long n;
        n = (long)FEAT * FEAT;
        k_wt_bf16<<<cdiv(n, TB), TB, 0, stream>>>((const float*)d_in[base + 0], wb,           FEAT, FEAT);
        k_wt_bf16<<<cdiv(n, TB), TB, 0, stream>>>((const float*)d_in[base + 2], wb + 262144,  FEAT, FEAT);
        n = (long)2 * FEAT * FEAT;
        k_wt_bf16<<<cdiv(n, TB), TB, 0, stream>>>((const float*)d_in[base + 4], wb + 524288, 2 * FEAT, FEAT);
    }

    // one hetero-SAGE block
    auto sage = [&](const __bf16* srcbf, const __bf16* dstbf, const int* ei,
                    int nDst, int s, void* outPtr, int outMode) {
        const float* bn = (const float*)d_in[4 + s * 6 + 1];
        const float* bs = (const float*)d_in[4 + s * 6 + 3];
        const float* bu = (const float*)d_in[4 + s * 6 + 5];
        const __bf16* WnT = WT + (size_t)s * WBLK;
        const __bf16* WsT = WnT + 262144;
        const __bf16* WuT = WnT + 524288;

        long n4 = (long)nDst * FEAT / 4;
        k_zero_f32<<<cdiv(n4, TB), TB, 0, stream>>>(aggf, n4);
        k_zero_f32<<<cdiv((long)cdiv(nDst, 4), 1) /*rows/4*/ ? cdiv((long)nDst / 4 + 1, TB) : 1,
                    TB, 0, stream>>>(cnt, (long)(nDst + 3) / 4);
        k_count<<<cdiv(E, TB), TB, 0, stream>>>(ei + E, cnt, E);
        long nsc = (long)E * (FEAT / 8);
        k_scatter<<<cdiv(nsc, TB), TB, 0, stream>>>(srcbf, ei, aggf, E);
        long nn = (long)nDst * (FEAT / 8);
        k_norm_bf16<<<cdiv(nn, TB), TB, 0, stream>>>(aggf, cnt, aggbf, nDst);

        dim3 g(cdiv(nDst, TM), FEAT / TN);
        // hcat[:, 0:512]   = aggbf @ Wn + bn   (bf16)
        k_gemm_wmma<1><<<g, TB, 0, stream>>>(aggbf, WnT, bn, hcat, nDst, FEAT, 2 * FEAT, 0);
        // hcat[:, 512:1024] = dstbf @ Ws + bs  (bf16)
        k_gemm_wmma<1><<<g, TB, 0, stream>>>(dstbf, WsT, bs, hcat, nDst, FEAT, 2 * FEAT, FEAT);
        // out = hcat @ Wu + bu
        if (outMode == 2)
            k_gemm_wmma<2><<<g, TB, 0, stream>>>(hcat, WuT, bu, outPtr, nDst, 2 * FEAT, FEAT, 0);
        else
            k_gemm_wmma<0><<<g, TB, 0, stream>>>(hcat, WuT, bu, outPtr, nDst, 2 * FEAT, FEAT, 0);
    };

    float* o0 = (float*)d_out;
    float* o1 = (float*)d_out + (size_t)N0 * FEAT;

    // layer 1 (fused lrelu+bf16 epilogue -> layer-2 inputs)
    sage(hb0, hb1, ei01, N1, 0, hb1n, 2);   // o1 = sage(h0 -> n1)
    sage(hb1, hb0, ei10, N0, 1, hb0n, 2);   // o0 = sage(h1 -> n0)
    // layer 2 (f32 outputs)
    sage(hb0n, hb1n, ei01, N1, 2, o1, 0);
    sage(hb1n, hb0n, ei10, N0, 3, o0, 0);
}
